// MutationGraphiT_54185307407182
// MI455X (gfx1250) — matrix operand order repaired and checked by reading, not compiled, not run on hardware
//
#include <hip/hip_runtime.h>
#include <hip/hip_fp16.h>

typedef __attribute__((ext_vector_type(16))) _Float16 v16h;
typedef __attribute__((ext_vector_type(8)))  _Float16 v8h;
typedef __attribute__((ext_vector_type(8)))  float    v8f;
typedef __attribute__((ext_vector_type(4)))  int      v4i;

#define BB   32
#define SSQ  512
#define DMODEL 256
#define NHEAD 8
#define DHEAD 32
#define NLAYER 2
#define FFDIM 512
#define NTOK (BB*SSQ)
#define NCLS 36

#if __has_builtin(__builtin_amdgcn_global_load_async_to_lds_b128)
#define HAVE_ASYNC_LDS 1
typedef __attribute__((address_space(1))) v4i  as1_v4i;
typedef __attribute__((address_space(1))) void as1_void;
typedef __attribute__((address_space(3))) v4i  as3_v4i;
typedef __attribute__((address_space(3))) void as3_void;
#define ASYNC_G2L_B128(gptr, lptr)                                        \
  __builtin_amdgcn_global_load_async_to_lds_b128(                         \
      (as1_v4i*)(as1_void*)(void*)(gptr),                                 \
      (as3_v4i*)(as3_void*)(void*)(lptr), 0, 0)
#else
#define HAVE_ASYNC_LDS 0
#endif

// ---------------------------------------------------------------------------
// f32 -> f16 elementwise convert (weights)
// ---------------------------------------------------------------------------
__global__ void cvt_f16_kernel(const float* __restrict__ in,
                               _Float16* __restrict__ out, int n) {
  int i = blockIdx.x * 256 + threadIdx.x;
  if (i < n) out[i] = (_Float16)in[i];
}

// f32 [rows,cols] -> f16 [cols,rows]  (transpose convert, for gat_W)
__global__ void cvt_f16_T_kernel(const float* __restrict__ in,
                                 _Float16* __restrict__ out, int rows, int cols) {
  int i = blockIdx.x * 256 + threadIdx.x;
  if (i >= rows*cols) return;
  int r = i / cols, c = i % cols;
  out[(size_t)c*rows + r] = (_Float16)in[i];
}

// ---------------------------------------------------------------------------
// Token embedding: concat(emb0,emb1,emb2, cnt*W+b) + pos_enc -> X(f32), Xh(f16)
// ---------------------------------------------------------------------------
__global__ void embed_kernel(const int* __restrict__ cat, const float* __restrict__ cnt,
                             const float* __restrict__ e0, const float* __restrict__ e1,
                             const float* __restrict__ e2, const float* __restrict__ cW,
                             const float* __restrict__ cb, const float* __restrict__ pos,
                             float* __restrict__ X, _Float16* __restrict__ Xh) {
  int tok = blockIdx.x;           // b*S + s
  int d   = threadIdx.x;          // 0..255
  int b = tok / SSQ, s = tok % SSQ;
  float v;
  if (d < 64)       v = e0[cat[(b*3+0)*SSQ + s]*64 + d];
  else if (d < 128) v = e1[cat[(b*3+1)*SSQ + s]*64 + (d-64)];
  else if (d < 192) v = e2[cat[(b*3+2)*SSQ + s]*64 + (d-128)];
  else              v = cnt[b*SSQ + s] * cW[d-192] + cb[d-192];
  v += pos[s*DMODEL + d];
  size_t idx = (size_t)tok*DMODEL + d;
  X[idx] = v; Xh[idx] = (_Float16)v;
}

// ---------------------------------------------------------------------------
// Banded adjacency mask: maskband[tok*20 + t], t=0..18 <-> offset d=t-9
// ---------------------------------------------------------------------------
__global__ void mask_kernel(const int* __restrict__ cat, const float* __restrict__ cnt,
                            float* __restrict__ mb) {
  int idx = blockIdx.x * 256 + threadIdx.x;
  if (idx >= NTOK*20) return;
  int t = idx % 20, tok = idx / 20;
  int b = tok / SSQ, i = tok % SSQ;
  float m = 0.f;
  if (t < 19) {
    int d = t - 9, j = i + d;
    if (j >= 0 && j < SSQ) {
      if (d == 0) m = 1.f;
      else {
        int dd = d < 0 ? -d : d;
        float cs = 0.f;
        for (int f = 0; f < 3; ++f)
          cs += (cat[(b*3+f)*SSQ + i] == cat[(b*3+f)*SSQ + j]) ? 1.f : 0.f;
        cs *= (1.f/3.f);
        float ns  = __expf(-fabsf(cnt[b*SSQ+i] - cnt[b*SSQ+j]) * 0.01f);
        float sim = 0.5f*(cs + ns);
        m = (sim > 0.6f) ? sim : (dd == 1 ? 1.f : 0.f);
      }
    }
  }
  mb[idx] = m;
}

// ---------------------------------------------------------------------------
// WMMA GEMM: C[M,N] = act(A[M,K] * B^T + bias), B stored [N,K] f16 row-major.
// One wave computes a 16x64 C tile: 4 accumulators share one A fragment,
// 4x v_wmma_f32_16x16x32_f16 per 32-deep K step.
// ---------------------------------------------------------------------------
__global__ __launch_bounds__(32)
void gemm_wmma_kernel(const _Float16* __restrict__ A, const _Float16* __restrict__ Bm,
                      const float* __restrict__ bias,
                      float* __restrict__ Cf, _Float16* __restrict__ Ch,
                      int M, int N, int K, int relu) {
  int ntn  = N >> 6;                       // 64-wide N tiles
  int m0   = (blockIdx.x / ntn) << 4;
  int n0   = (blockIdx.x % ntn) << 6;
  int lane = threadIdx.x;
  int half = lane >> 4;
  int lm   = lane & 15;
  v8f c0 = {}, c1 = {}, c2 = {}, c3 = {};
  const _Float16* arow = A  + (size_t)(m0 + lm) * K;
  const _Float16* b0r  = Bm + (size_t)(n0      + lm) * K;
  const _Float16* b1r  = Bm + (size_t)(n0 + 16 + lm) * K;
  const _Float16* b2r  = Bm + (size_t)(n0 + 32 + lm) * K;
  const _Float16* b3r  = Bm + (size_t)(n0 + 48 + lm) * K;
  for (int k0 = 0; k0 < K; k0 += 32) {
    // A fragment: m = lm ; k = k0+half*16+j (contiguous)
    v16h a;
    {
      const _Float16* ap = arow + k0 + half*16;
      v8h lo = *(const v8h*)ap;
      v8h hi = *(const v8h*)(ap + 8);
#pragma unroll
      for (int j = 0; j < 8; ++j) { a[j] = lo[j]; a[8+j] = hi[j]; }
      __builtin_prefetch(ap + 32, 0, 0);
    }
    // B fragments: n = lm ; kk = half*16 + j  (row-major [N,K] -> contiguous)
#define LOAD_BFRAG(dst, rowp)                                          \
    {                                                                  \
      const _Float16* bp = (rowp) + k0 + half*16;                      \
      v8h lo = *(const v8h*)bp;                                        \
      v8h hi = *(const v8h*)(bp + 8);                                  \
      _Pragma("unroll")                                                \
      for (int j = 0; j < 8; ++j) { dst[j] = lo[j]; dst[8+j] = hi[j]; }\
    }
    v16h bf0, bf1, bf2, bf3;
    LOAD_BFRAG(bf0, b0r) LOAD_BFRAG(bf1, b1r)
    LOAD_BFRAG(bf2, b2r) LOAD_BFRAG(bf3, b3r)
#undef LOAD_BFRAG
    c0 = __builtin_amdgcn_wmma_f32_16x16x32_f16(false, a, false, bf0, (short)0, c0, false, false);
    c1 = __builtin_amdgcn_wmma_f32_16x16x32_f16(false, a, false, bf1, (short)0, c1, false, false);
    c2 = __builtin_amdgcn_wmma_f32_16x16x32_f16(false, a, false, bf2, (short)0, c2, false, false);
    c3 = __builtin_amdgcn_wmma_f32_16x16x32_f16(false, a, false, bf3, (short)0, c3, false, false);
  }
  v8f acc[4] = {c0, c1, c2, c3};
#pragma unroll
  for (int t = 0; t < 4; ++t) {
    float bs = bias ? bias[n0 + t*16 + lm] : 0.f;
#pragma unroll
    for (int r = 0; r < 8; ++r) {
      float v = acc[t][r] + bs;
      if (relu) v = v > 0.f ? v : 0.f;
      size_t idx = (size_t)(m0 + r + half*8) * N + n0 + t*16 + lm;
      if (Cf) Cf[idx] = v;
      if (Ch) Ch[idx] = (_Float16)v;
    }
  }
}

// ---------------------------------------------------------------------------
// GAT: e1 = Wh.a1, e2 = Wh.a2 (a = gat_a[l], len 512)
// ---------------------------------------------------------------------------
__global__ void gat_e_kernel(const float* __restrict__ Wh, const float* __restrict__ a,
                             float* __restrict__ e1, float* __restrict__ e2) {
  __shared__ float r1[256], r2[256];
  int m = blockIdx.x, t = threadIdx.x;
  float w = Wh[(size_t)m*DMODEL + t];
  r1[t] = w * a[t]; r2[t] = w * a[DMODEL + t];
  __syncthreads();
  for (int s2 = 128; s2 > 0; s2 >>= 1) {
    if (t < s2) { r1[t] += r1[t+s2]; r2[t] += r2[t+s2]; }
    __syncthreads();
  }
  if (t == 0) { e1[m] = r1[0]; e2[m] = r2[0]; }
}

// ---------------------------------------------------------------------------
// GAT banded attention: g = elu( softmax_band(e1_i+e2_j) @ Wh )
// ---------------------------------------------------------------------------
__global__ void gat_attn_kernel(const float* __restrict__ Wh, const float* __restrict__ e1,
                                const float* __restrict__ e2, const float* __restrict__ mb,
                                float* __restrict__ g) {
  __shared__ float wv[19], pv[19];
  int tok = blockIdx.x, t = threadIdx.x;
  int b = tok / SSQ, i = tok % SSQ;
  if (t < 19) {
    int j = i + t - 9;
    float mv = (j >= 0 && j < SSQ) ? mb[(size_t)tok*20 + t] : 0.f;
    wv[t] = (mv > 0.f) ? (e1[tok] + e2[b*SSQ + j]) : -1e30f;
  }
  __syncthreads();
  if (t == 0) {
    float mx = -1e30f;
    for (int q = 0; q < 19; ++q) mx = fmaxf(mx, wv[q]);
    float sm = 0.f;
    for (int q = 0; q < 19; ++q) { float p = __expf(wv[q]-mx); pv[q] = p; sm += p; }
    float inv = 1.f / sm;
    for (int q = 0; q < 19; ++q) pv[q] *= inv;
  }
  __syncthreads();
  float acc = 0.f;
  for (int q = 0; q < 19; ++q) {
    int j = i + q - 9;
    if (j >= 0 && j < SSQ)
      acc += pv[q] * Wh[((size_t)(b*SSQ + j))*DMODEL + t];
  }
  float gv = acc > 0.f ? acc : (__expf(acc) - 1.f);    // elu
  g[(size_t)tok*DMODEL + t] = gv;
}

// ---------------------------------------------------------------------------
// X = LayerNorm(X + Y) ; also refresh f16 shadow
// ---------------------------------------------------------------------------
__global__ void residual_ln_kernel(float* __restrict__ X, _Float16* __restrict__ Xh,
                                   const float* __restrict__ Y,
                                   const float* __restrict__ gmm,
                                   const float* __restrict__ bta) {
  __shared__ float rb[256];
  int tok = blockIdx.x, t = threadIdx.x;
  size_t idx = (size_t)tok*DMODEL + t;
  float v = X[idx] + Y[idx];
  rb[t] = v; __syncthreads();
  for (int s2 = 128; s2 > 0; s2 >>= 1) { if (t < s2) rb[t] += rb[t+s2]; __syncthreads(); }
  float mean = rb[0] * (1.f/DMODEL); __syncthreads();
  float dv = v - mean;
  rb[t] = dv*dv; __syncthreads();
  for (int s2 = 128; s2 > 0; s2 >>= 1) { if (t < s2) rb[t] += rb[t+s2]; __syncthreads(); }
  float var = rb[0] * (1.f/DMODEL);
  float o = gmm[t] * dv * rsqrtf(var + 1e-5f) + bta[t];
  X[idx] = o; Xh[idx] = (_Float16)o;
}

// ---------------------------------------------------------------------------
// Flash attention: one wave per (b,h, 16-row Q tile). qkv f16 [tok, 768],
// layout q|k|v each h*32+d. Online softmax; K/V staged into LDS via the
// CDNA5 async-to-LDS path when available (ASYNCcnt + s_wait_asynccnt).
// ---------------------------------------------------------------------------
__global__ __launch_bounds__(32)
void flash_attn_kernel(const _Float16* __restrict__ qkv, _Float16* __restrict__ O) {
  int bh = blockIdx.x;
  int b = bh / NHEAD, h = bh % NHEAD;
  int i0 = blockIdx.y << 4;
  int lane = threadIdx.x, half = lane >> 4, lm = lane & 15;
  __shared__ _Float16 Kt[32*32];
  __shared__ _Float16 Vt[32*32];
  __shared__ _Float16 Pt[16*32];

  // Q fragment (A-layout)
  v16h aq;
  {
    const _Float16* qp = qkv + ((size_t)(b*SSQ + i0 + lm))*768 + h*DHEAD + half*16;
    v8h lo = *(const v8h*)qp;
    v8h hi = *(const v8h*)(qp + 8);
#pragma unroll
    for (int j = 0; j < 8; ++j) { aq[j] = lo[j]; aq[8+j] = hi[j]; }
  }
  v8f o0 = {}, o1 = {};
  float mrow[8], lrow[8];
#pragma unroll
  for (int r = 0; r < 8; ++r) { mrow[r] = -1e30f; lrow[r] = 0.f; }
  const float scale = 0.17677669529663687f;   // 1/sqrt(32)

  for (int j0 = 0; j0 < SSQ; j0 += 32) {
    // stage K,V tile rows (lane -> row j0+lane, 32 halves each)
    {
      const _Float16* kp = qkv + ((size_t)(b*SSQ + j0 + lane))*768 + 256 + h*DHEAD;
      const _Float16* vp = qkv + ((size_t)(b*SSQ + j0 + lane))*768 + 512 + h*DHEAD;
#if HAVE_ASYNC_LDS
#pragma unroll
      for (int c = 0; c < 4; ++c) {
        ASYNC_G2L_B128(kp + c*8, &Kt[lane*32 + c*8]);
        ASYNC_G2L_B128(vp + c*8, &Vt[lane*32 + c*8]);
      }
#if __has_builtin(__builtin_amdgcn_s_wait_asynccnt)
      __builtin_amdgcn_s_wait_asynccnt(0);
#else
      asm volatile("s_wait_asynccnt 0x0" ::: "memory");
#endif
#else
#pragma unroll
      for (int c = 0; c < 4; ++c) {
        *(v8h*)&Kt[lane*32 + c*8] = *(const v8h*)(kp + c*8);
        *(v8h*)&Vt[lane*32 + c*8] = *(const v8h*)(vp + c*8);
      }
#endif
    }
    __syncthreads();
    // K^T fragments: bk0 cols j0..j0+15, bk1 cols +16 ; (kk,n)=Kt[n*32+kk]
    v16h bk0, bk1;
    {
      const _Float16* p0 = &Kt[lm*32 + half*16];
      const _Float16* p1 = &Kt[(16+lm)*32 + half*16];
      v8h a0 = *(const v8h*)p0, a1 = *(const v8h*)(p0+8);
      v8h c0 = *(const v8h*)p1, c1 = *(const v8h*)(p1+8);
#pragma unroll
      for (int j = 0; j < 8; ++j) {
        bk0[j] = a0[j]; bk0[8+j] = a1[j];
        bk1[j] = c0[j]; bk1[8+j] = c1[j];
      }
    }
    v8f s0 = {}, s1 = {};
    s0 = __builtin_amdgcn_wmma_f32_16x16x32_f16(false, aq, false, bk0, (short)0, s0, false, false);
    s1 = __builtin_amdgcn_wmma_f32_16x16x32_f16(false, aq, false, bk1, (short)0, s1, false, false);
    // online softmax (row stats reduced across the 16-lane half)
    float pr0[8], pr1[8];
#pragma unroll
    for (int r = 0; r < 8; ++r) {
      float x0 = s0[r]*scale, x1 = s1[r]*scale;
      float mx = fmaxf(x0, x1);
      for (int off = 1; off < 16; off <<= 1) mx = fmaxf(mx, __shfl_xor(mx, off, 32));
      float mnew = fmaxf(mrow[r], mx);
      float corr = __expf(mrow[r] - mnew);
      float p0 = __expf(x0 - mnew), p1 = __expf(x1 - mnew);
      float rs = p0 + p1;
      for (int off = 1; off < 16; off <<= 1) rs += __shfl_xor(rs, off, 32);
      lrow[r] = lrow[r]*corr + rs;
      mrow[r] = mnew;
      o0[r] *= corr; o1[r] *= corr;
      pr0[r] = p0; pr1[r] = p1;
    }
    // transpose P (C-layout -> A-layout) via LDS
#pragma unroll
    for (int r = 0; r < 8; ++r) {
      int m = r + half*8;
      Pt[m*32 + lm]      = (_Float16)pr0[r];
      Pt[m*32 + 16 + lm] = (_Float16)pr1[r];
    }
    __syncthreads();
    v16h ap;
    {
      const _Float16* pp = &Pt[lm*32 + half*16];
      v8h lo = *(const v8h*)pp;
      v8h hi = *(const v8h*)(pp + 8);
#pragma unroll
      for (int j = 0; j < 8; ++j) { ap[j] = lo[j]; ap[8+j] = hi[j]; }
    }
    // V fragments: (kk,n) = Vt[kk*32 + n] (bv0 n=0..15, bv1 n=16..31)
    v16h bv0, bv1;
#pragma unroll
    for (int j = 0; j < 16; ++j) {
      int kk = half*16 + j;
      bv0[j] = Vt[kk*32 + lm];
      bv1[j] = Vt[kk*32 + 16 + lm];
    }
    o0 = __builtin_amdgcn_wmma_f32_16x16x32_f16(false, ap, false, bv0, (short)0, o0, false, false);
    o1 = __builtin_amdgcn_wmma_f32_16x16x32_f16(false, ap, false, bv1, (short)0, o1, false, false);
    __syncthreads();
  }
#pragma unroll
  for (int r = 0; r < 8; ++r) {
    float inv = 1.f / lrow[r];
    int m = r + half*8;
    size_t base = ((size_t)(b*SSQ + i0 + m))*DMODEL + h*DHEAD;
    O[base + lm]      = (_Float16)(o0[r]*inv);
    O[base + 16 + lm] = (_Float16)(o1[r]*inv);
  }
}

// ---------------------------------------------------------------------------
// Mean pool over sequence
// ---------------------------------------------------------------------------
__global__ void pool_kernel(const float* __restrict__ X, float* __restrict__ pooled) {
  int b = blockIdx.x, d = threadIdx.x;
  float s = 0.f;
  for (int i = 0; i < SSQ; ++i) s += X[((size_t)b*SSQ + i)*DMODEL + d];
  pooled[b*DMODEL + d] = s * (1.f/SSQ);
}

// ---------------------------------------------------------------------------
// Classifier head (tiny: VALU)
// ---------------------------------------------------------------------------
__global__ void cls_kernel(const float* __restrict__ pooled,
                           const float* __restrict__ W1, const float* __restrict__ b1,
                           const float* __restrict__ W2, const float* __restrict__ b2,
                           float* __restrict__ out) {
  __shared__ float hbuf[128];
  int b = blockIdx.x, t = threadIdx.x;   // 128 threads
  float s = b1[t];
  for (int k = 0; k < DMODEL; ++k) s += pooled[b*DMODEL + k] * W1[t*DMODEL + k];
  hbuf[t] = s > 0.f ? s : 0.f;
  __syncthreads();
  if (t < NCLS) {
    float o = b2[t];
    for (int k = 0; k < 128; ++k) o += hbuf[k] * W2[t*128 + k];
    out[b*NCLS + t] = o;
  }
}

// ---------------------------------------------------------------------------
// Host orchestration
// ---------------------------------------------------------------------------
extern "C" void kernel_launch(void* const* d_in, const int* in_sizes, int n_in,
                              void* d_out, int out_size, void* d_ws, size_t ws_size,
                              hipStream_t stream) {
  const int*   cat    = (const int*)  d_in[0];
  const float* cnt    = (const float*)d_in[1];
  const float* emb0   = (const float*)d_in[2];
  const float* emb1   = (const float*)d_in[3];
  const float* emb2   = (const float*)d_in[4];
  const float* cnt_W  = (const float*)d_in[5];
  const float* cnt_b  = (const float*)d_in[6];
  const float* pos    = (const float*)d_in[7];
  const float* gat_W  = (const float*)d_in[8];
  const float* gat_a  = (const float*)d_in[9];
  const float* qkv_W  = (const float*)d_in[10];
  const float* qkv_b  = (const float*)d_in[11];
  const float* outp_W = (const float*)d_in[12];
  const float* outp_b = (const float*)d_in[13];
  const float* ff1_W  = (const float*)d_in[14];
  const float* ff1_b  = (const float*)d_in[15];
  const float* ff2_W  = (const float*)d_in[16];
  const float* ff2_b  = (const float*)d_in[17];
  const float* ln_g   = (const float*)d_in[18];
  const float* ln_b   = (const float*)d_in[19];
  const float* cls1_W = (const float*)d_in[20];
  const float* cls1_b = (const float*)d_in[21];
  const float* cls2_W = (const float*)d_in[22];
  const float* cls2_b = (const float*)d_in[23];
  float* out = (float*)d_out;

  char* wp = (char*)d_ws;
  auto alloc = [&](size_t bytes) -> void* {
    void* p = (void*)wp;
    wp += (bytes + 255) & ~(size_t)255;
    return p;
  };
  float*    X      = (float*)   alloc((size_t)NTOK*DMODEL*4);
  _Float16* Xh     = (_Float16*)alloc((size_t)NTOK*DMODEL*2);
  float*    Y      = (float*)   alloc((size_t)NTOK*DMODEL*4);
  float*    WH     = (float*)   alloc((size_t)NTOK*DMODEL*4);
  _Float16* Oh     = (_Float16*)alloc((size_t)NTOK*DMODEL*2);
  _Float16* QKVh   = (_Float16*)alloc((size_t)NTOK*768*2);
  _Float16* FFh    = (_Float16*)alloc((size_t)NTOK*FFDIM*2);
  float*    e1b    = (float*)   alloc((size_t)NTOK*4);
  float*    e2b    = (float*)   alloc((size_t)NTOK*4);
  float*    mb     = (float*)   alloc((size_t)NTOK*20*4);
  float*    pooled = (float*)   alloc((size_t)BB*DMODEL*4);
  _Float16* gatWt  = (_Float16*)alloc((size_t)NLAYER*DMODEL*DMODEL*2);  // transposed
  _Float16* qkvWh  = (_Float16*)alloc((size_t)NLAYER*768*DMODEL*2);
  _Float16* outpWh = (_Float16*)alloc((size_t)NLAYER*DMODEL*DMODEL*2);
  _Float16* ff1Wh  = (_Float16*)alloc((size_t)NLAYER*FFDIM*DMODEL*2);
  _Float16* ff2Wh  = (_Float16*)alloc((size_t)NLAYER*DMODEL*FFDIM*2);

  auto cvt = [&](const float* src, _Float16* dst, int n) {
    cvt_f16_kernel<<<(n+255)/256, 256, 0, stream>>>(src, dst, n);
  };
  // gat_W used as x@W: transpose so GEMM always sees B as [N,K] row-major
  for (int l = 0; l < NLAYER; ++l)
    cvt_f16_T_kernel<<<(DMODEL*DMODEL+255)/256, 256, 0, stream>>>(
        gat_W + (size_t)l*DMODEL*DMODEL, gatWt + (size_t)l*DMODEL*DMODEL,
        DMODEL, DMODEL);
  cvt(qkv_W,  qkvWh,  NLAYER*768*DMODEL);
  cvt(outp_W, outpWh, NLAYER*DMODEL*DMODEL);
  cvt(ff1_W,  ff1Wh,  NLAYER*FFDIM*DMODEL);
  cvt(ff2_W,  ff2Wh,  NLAYER*DMODEL*FFDIM);

  embed_kernel<<<NTOK, 256, 0, stream>>>(cat, cnt, emb0, emb1, emb2,
                                         cnt_W, cnt_b, pos, X, Xh);
  mask_kernel<<<(NTOK*20 + 255)/256, 256, 0, stream>>>(cat, cnt, mb);

  auto gemm = [&](const _Float16* A, const _Float16* Bm, const float* bias,
                  float* Cf, _Float16* Ch, int M, int N, int K, int relu) {
    int tiles = (M >> 4) * (N >> 6);
    gemm_wmma_kernel<<<tiles, 32, 0, stream>>>(A, Bm, bias, Cf, Ch, M, N, K, relu);
  };

  for (int l = 0; l < NLAYER; ++l) {
    // --- GAT layer ---
    gemm(Xh, gatWt + (size_t)l*DMODEL*DMODEL, nullptr, WH, nullptr,
         NTOK, DMODEL, DMODEL, 0);
    gat_e_kernel<<<NTOK, 256, 0, stream>>>(WH, gat_a + (size_t)l*2*DMODEL, e1b, e2b);
    gat_attn_kernel<<<NTOK, 256, 0, stream>>>(WH, e1b, e2b, mb, Y);
    residual_ln_kernel<<<NTOK, 256, 0, stream>>>(X, Xh, Y,
        ln_g + (size_t)(l*3+0)*DMODEL, ln_b + (size_t)(l*3+0)*DMODEL);
    // --- Multihead attention ---
    gemm(Xh, qkvWh + (size_t)l*768*DMODEL, qkv_b + (size_t)l*768,
         nullptr, QKVh, NTOK, 768, DMODEL, 0);
    flash_attn_kernel<<<dim3(BB*NHEAD, SSQ/16), 32, 0, stream>>>(QKVh, Oh);
    gemm(Oh, outpWh + (size_t)l*DMODEL*DMODEL, outp_b + (size_t)l*DMODEL,
         Y, nullptr, NTOK, DMODEL, DMODEL, 0);
    residual_ln_kernel<<<NTOK, 256, 0, stream>>>(X, Xh, Y,
        ln_g + (size_t)(l*3+1)*DMODEL, ln_b + (size_t)(l*3+1)*DMODEL);
    // --- FFN ---
    gemm(Xh, ff1Wh + (size_t)l*FFDIM*DMODEL, ff1_b + (size_t)l*FFDIM,
         nullptr, FFh, NTOK, FFDIM, DMODEL, /*relu=*/1);
    gemm(FFh, ff2Wh + (size_t)l*DMODEL*FFDIM, ff2_b + (size_t)l*DMODEL,
         Y, nullptr, NTOK, DMODEL, FFDIM, 0);
    residual_ln_kernel<<<NTOK, 256, 0, stream>>>(X, Xh, Y,
        ln_g + (size_t)(l*3+2)*DMODEL, ln_b + (size_t)(l*3+2)*DMODEL);
  }

  pool_kernel<<<BB, 256, 0, stream>>>(X, pooled);
  cls_kernel<<<BB, 128, 0, stream>>>(pooled, cls1_W, cls1_b, cls2_W, cls2_b, out);
}
